// MultiHeadSelfAttention_83665962926361
// MI455X (gfx1250) — matrix operand order repaired and checked by reading, not compiled, not run on hardware
//
#include <hip/hip_runtime.h>
#include <hip/hip_bf16.h>

#define NUM_B 4
#define NH    8
#define SEQ   4096
#define CDIM  256
#define DH    32
#define MTOT  (NUM_B*SEQ)

#define QSCALE (0.1767766953f * 1.4426950409f)  // 1/sqrt(dh) * log2(e), folded into Q

typedef __attribute__((ext_vector_type(16))) __bf16 v16bf;
typedef __attribute__((ext_vector_type(8)))  float  v8f;

#ifndef __has_builtin
#define __has_builtin(x) 0
#endif
#if defined(__AMDGCN__) && __has_builtin(__builtin_amdgcn_tensor_load_to_lds) && \
    __has_builtin(__builtin_amdgcn_s_wait_tensorcnt)
#define HAS_TDM 1
#else
#define HAS_TDM 0
#endif

union AFrag { v16bf v; __bf16 e[16]; uint4 u[2]; };

static __device__ __forceinline__ v16bf ld_frag2(const __bf16* p0, const __bf16* p1) {
  AFrag f;
  f.u[0] = *reinterpret_cast<const uint4*>(p0);
  f.u[1] = *reinterpret_cast<const uint4*>(p1);
  return f.v;
}
static __device__ __forceinline__ v16bf ld_frag_lds(const __bf16* p) {
  return ld_frag2(p, p + 8);
}
static __device__ __forceinline__ v16bf ld_frag_f32(const float* p0, const float* p1) {
  AFrag f;
#pragma unroll
  for (int i = 0; i < 8; ++i) f.e[i]     = (__bf16)p0[i];
#pragma unroll
  for (int i = 0; i < 8; ++i) f.e[8 + i] = (__bf16)p1[i];
  return f.v;
}
static __device__ __forceinline__ v8f wmma_bf16(v16bf a, v16bf b, v8f c) {
  return __builtin_amdgcn_wmma_f32_16x16x32_bf16(false, a, false, b, (short)0, c, false, false);
}

#if HAS_TDM
typedef unsigned int v4u __attribute__((ext_vector_type(4)));
typedef int          v8i __attribute__((ext_vector_type(8)));
typedef int          v4i __attribute__((ext_vector_type(4)));

// 2D tile DMA: 32 rows x 32 bf16 elements, row pitch `dim0_stride` elements.
static __device__ __forceinline__ void tdm_load_2d(unsigned int lds_addr,
                                                   unsigned long long gaddr,
                                                   unsigned int dim0_stride) {
  v4u g0;
  g0[0] = 1u;                                            // count = 1 valid descriptor
  g0[1] = lds_addr;                                      // LDS byte address
  g0[2] = (unsigned int)(gaddr & 0xFFFFFFFFull);         // global addr [31:0]
  g0[3] = (unsigned int)((gaddr >> 32) & 0x1FFFFFFull)   // global addr [56:32]
        | (2u << 30);                                    // type = 2 ("image")
  v8i g1;
  g1[0] = 0x00010000;                 // wg_mask=0, data_size=1 (2 bytes)
  g1[1] = (int)(32u << 16);           // tensor_dim0 = 32 (bits 79:48, low half)
  g1[2] = (int)(4096u << 16);         // tensor_dim1 = 4096 (bits 111:80, low half)
  g1[3] = (int)(32u << 16);           // tile_dim0 = 32 (bits 127:112)
  g1[4] = 32;                         // tile_dim1 = 32 (bits 143:128); tile_dim2 = 0
  g1[5] = (int)dim0_stride;           // tensor_dim0_stride (bits 207:160, low 32)
  g1[6] = 0;                          // stride hi / dim1_stride lo
  g1[7] = 0;                          // dim1_stride hi
  v4i z4 = {0, 0, 0, 0};
#if __clang_major__ >= 23
  v8i z8 = {0, 0, 0, 0, 0, 0, 0, 0};
  __builtin_amdgcn_tensor_load_to_lds(g0, g1, z4, z4, z8, 0);
#else
  __builtin_amdgcn_tensor_load_to_lds(g0, g1, z4, z4, 0);
#endif
}
#endif

// ---------------------------------------------------------------------------
// Kernel 1: fused QKV projection  y = x @ W^T + b  (one weight per blockIdx.y)
// Q (pre-scaled by QSCALE) / K out: [b, h, n, 32] bf16.
// V out: [b*n, 256] bf16 (residual) AND transposed [b, h, d, n] bf16 (for PV).
// ---------------------------------------------------------------------------
__global__ __launch_bounds__(256)
void qkv_proj_kernel(const float* __restrict__ x,
                     const float* __restrict__ Wq, const float* __restrict__ bq,
                     const float* __restrict__ Wk, const float* __restrict__ bk,
                     const float* __restrict__ Wv, const float* __restrict__ bv,
                     __bf16* __restrict__ qbuf, __bf16* __restrict__ kbuf,
                     __bf16* __restrict__ vbuf, __bf16* __restrict__ vtbuf)
{
  __shared__ __bf16 Wsl[CDIM * 32];               // W[:, ks:ks+32] as bf16 (16 KB)
  const int m = blockIdx.y;
  const float* Wm = (m == 0) ? Wq : ((m == 1) ? Wk : Wv);
  const float* bm = (m == 0) ? bq : ((m == 1) ? bk : bv);
  const int tid  = threadIdx.x;
  const int lane = tid & 31, wave = tid >> 5;
  const int r0   = blockIdx.x * 128 + wave * 16;  // 16 output rows per wave
  const int col  = lane & 15;
  const bool hi  = lane >= 16;

  v8f acc[16] = {};
  for (int ks = 0; ks < CDIM; ks += 32) {
    __syncthreads();
    { // cooperative stage: thread tid owns row n = tid of W slice
      const float* src = Wm + (size_t)tid * CDIM + ks;
      __bf16 tmp[32];
#pragma unroll
      for (int i = 0; i < 32; ++i) tmp[i] = (__bf16)src[i];
      uint4* dst = reinterpret_cast<uint4*>(&Wsl[tid * 32]);
      const uint4* s4 = reinterpret_cast<const uint4*>(tmp);
#pragma unroll
      for (int i = 0; i < 4; ++i) dst[i] = s4[i];
    }
    __syncthreads();
    // A fragment per ISA 16-bit A layout (lane<16: K 0-7,16-23; lane>=16: +8)
    const int kofs = ks + (hi ? 8 : 0);
    const float* xp = x + (size_t)(r0 + col) * CDIM + kofs;
    const v16bf a = ld_frag_f32(xp, xp + 16);
#pragma unroll
    for (int t = 0; t < 16; ++t) {
      const __bf16* wp = &Wsl[(t * 16 + col) * 32 + (hi ? 8 : 0)];
      acc[t] = wmma_bf16(a, ld_frag2(wp, wp + 16), acc[t]);
    }
  }
  // bias + store (C-layout: row = j + 8*hi, col = t*16 + lane%16)
#pragma unroll
  for (int t = 0; t < 16; ++t) {
    const int c = t * 16 + col;
    const float bias = bm[c];
#pragma unroll
    for (int j = 0; j < 8; ++j) {
      const int rg = r0 + j + (hi ? 8 : 0);
      float fv = acc[t][j] + bias;
      if (m == 0) fv *= QSCALE;
      const __bf16 val = (__bf16)fv;
      const int b = rg >> 12, n = rg & (SEQ - 1);
      const int h = c >> 5, d = c & 31;
      if (m < 2) {
        __bf16* dst = (m == 0) ? qbuf : kbuf;
        dst[((size_t)(b * NH + h) * SEQ + n) * DH + d] = val;
      } else {
        vbuf[(size_t)rg * CDIM + c] = val;
        vtbuf[((size_t)(b * NH + h) * DH + d) * SEQ + n] = val;
      }
    }
  }
}

// ---------------------------------------------------------------------------
// Kernel 2: flash attention per (b, h, 128-row q tile), 32-key steps.
// K/V tiles double-buffered in LDS; staged by the Tensor Data Mover when
// available (wave 0 issues, TENSORcnt-pipelined), manual copy otherwise.
// ---------------------------------------------------------------------------
__global__ __launch_bounds__(256)
void flash_attn_kernel(const __bf16* __restrict__ qbuf,
                       const __bf16* __restrict__ kbuf,
                       const __bf16* __restrict__ vtbuf,
                       __bf16* __restrict__ aobuf)
{
  __shared__ __bf16 Kt[2][32 * 32];   // [key][d]
  __shared__ __bf16 Vt[2][32 * 32];   // [d][key] (pre-transposed in vtbuf)
  __shared__ __bf16 Pt[8][32 * 16];   // per-wave P tile in A-fragment layout
  const int b = blockIdx.z, h = blockIdx.y;
  const int bh = b * NH + h;
  const int tid  = threadIdx.x;
  const int lane = tid & 31, wave = tid >> 5;
  const int q0   = blockIdx.x * 128 + wave * 16;
  const int col  = lane & 15;
  const bool hi  = lane >= 16;

  // Q fragment (A layout, K dimension = dh = 32) -- loaded once, pre-scaled
  const int d0 = hi ? 8 : 0;
  const __bf16* qp = qbuf + ((size_t)bh * SEQ + q0 + col) * DH + d0;
  const v16bf qf = ld_frag2(qp, qp + 16);

  float mst[8], lsum[8];
#pragma unroll
  for (int j = 0; j < 8; ++j) { mst[j] = -1e30f; lsum[j] = 0.f; }
  v8f acc0 = {}, acc1 = {};
  __bf16* ptw = &Pt[wave][0];

  const size_t kbase = (size_t)bh * SEQ * DH;   // K rows: 32 bf16, pitch DH
  const size_t vbase = (size_t)bh * DH * SEQ;   // V^T rows: 32 bf16, pitch SEQ

  const int NIT = SEQ / 32;
#if HAS_TDM
  if (tid < 32) {
    tdm_load_2d((unsigned int)(uintptr_t)&Kt[0][0],
                (unsigned long long)(uintptr_t)(kbuf + kbase), DH);
    tdm_load_2d((unsigned int)(uintptr_t)&Vt[0][0],
                (unsigned long long)(uintptr_t)(vtbuf + vbase), SEQ);
  }
#endif

  for (int it = 0; it < NIT; ++it) {
    const int cur = it & 1;
    __syncthreads();   // prior compute on buf[!cur] finished -> safe to refill
#if HAS_TDM
    if (tid < 32) {
      if (it + 1 < NIT) {
        const int nxt = (it + 1) & 1;
        tdm_load_2d((unsigned int)(uintptr_t)&Kt[nxt][0],
                    (unsigned long long)(uintptr_t)(kbuf + kbase + (size_t)(it + 1) * 32 * DH), DH);
        tdm_load_2d((unsigned int)(uintptr_t)&Vt[nxt][0],
                    (unsigned long long)(uintptr_t)(vtbuf + vbase + (size_t)(it + 1) * 32), SEQ);
        __builtin_amdgcn_s_wait_tensorcnt(2);  // tile `it` complete, next in flight
      } else {
        __builtin_amdgcn_s_wait_tensorcnt(0);
      }
    }
#else
    { // manual staging: 8 bytes per thread per tile
      const int row = tid >> 3, e4 = (tid & 7) * 4;
      const int kv = it * 32;
      *reinterpret_cast<uint2*>(&Kt[cur][row * 32 + e4]) =
          *reinterpret_cast<const uint2*>(kbuf + kbase + (size_t)(kv + row) * DH + e4);
      *reinterpret_cast<uint2*>(&Vt[cur][row * 32 + e4]) =
          *reinterpret_cast<const uint2*>(vtbuf + vbase + (size_t)row * SEQ + kv + e4);
      __builtin_prefetch(kbuf + kbase + (size_t)(kv + 32 + row) * DH + e4, 0, 1);
      __builtin_prefetch(vtbuf + vbase + (size_t)row * SEQ + kv + 32 + e4, 0, 1);
    }
#endif
    __syncthreads();   // tile `cur` visible to all waves

    // scores: B[k=d][n=key] = K[key][d]; two 16-key subtiles (Q pre-scaled)
    const __bf16* kp0 = &Kt[cur][col * 32 + (hi ? 8 : 0)];
    const v8f z = {};
    v8f s0 = wmma_bf16(qf, ld_frag2(kp0, kp0 + 16), z);
    v8f s1 = wmma_bf16(qf, ld_frag2(kp0 + 16 * 32, kp0 + 16 * 32 + 16), z);

    // online softmax in exp2 domain; row r = j + 8*hi lives in this 16-lane group
#pragma unroll
    for (int j = 0; j < 8; ++j) {
      const float a = s0[j], bbv = s1[j];
      float rmax = fmaxf(a, bbv);
#pragma unroll
      for (int o = 1; o < 16; o <<= 1) rmax = fmaxf(rmax, __shfl_xor(rmax, o, 32));
      const float mnew  = fmaxf(mst[j], rmax);
      const float alpha = exp2f(mst[j] - mnew);
      mst[j] = mnew;
      const float p0 = exp2f(a - mnew), p1 = exp2f(bbv - mnew);
      float rs = p0 + p1;
#pragma unroll
      for (int o = 1; o < 16; o <<= 1) rs += __shfl_xor(rs, o, 32);
      lsum[j] = lsum[j] * alpha + rs;
      acc0[j] *= alpha;
      acc1[j] *= alpha;
      // scatter P from C-layout into A-fragment layout staging
      const int hb  = (col >= 8) ? 1 : 0;
      const int dl  = j + (hi ? 8 : 0) + 16 * hb;
      const int idx = col & 7;
      ptw[dl * 16 + idx]     = (__bf16)p0;
      ptw[dl * 16 + 8 + idx] = (__bf16)p1;
    }
    asm volatile("s_wait_dscnt 0" ::: "memory");
    const v16bf pf = ld_frag_lds(&ptw[lane * 16]);
    // PV: B[k=key][n=d] = V[key][d] -> contiguous rows of Vt (= V^T)
    const __bf16* vp0 = &Vt[cur][col * 32 + (hi ? 8 : 0)];
    acc0 = wmma_bf16(pf, ld_frag2(vp0, vp0 + 16), acc0);
    acc1 = wmma_bf16(pf, ld_frag2(vp0 + 16 * 32, vp0 + 16 * 32 + 16), acc1);
  }

  // normalize + store attention output [b*n, 256] bf16
#pragma unroll
  for (int j = 0; j < 8; ++j) {
    const float inv = 1.0f / lsum[j];
    const size_t rg = (size_t)(b * SEQ + q0 + j + (hi ? 8 : 0));
    aobuf[rg * CDIM + h * DH + col]      = (__bf16)(acc0[j] * inv);
    aobuf[rg * CDIM + h * DH + 16 + col] = (__bf16)(acc1[j] * inv);
  }
}

// ---------------------------------------------------------------------------
// Kernel 3: out-proj + bias + residual(V) + LayerNorm -> f32 output
// ---------------------------------------------------------------------------
__global__ __launch_bounds__(256)
void oproj_ln_kernel(const __bf16* __restrict__ aobuf,
                     const float* __restrict__ Wo, const float* __restrict__ bo,
                     const __bf16* __restrict__ vres,
                     const float* __restrict__ gamma, const float* __restrict__ beta,
                     float* __restrict__ out)
{
  __shared__ __bf16 Wsl[CDIM * 32];
  const int tid  = threadIdx.x;
  const int lane = tid & 31, wave = tid >> 5;
  const int r0   = blockIdx.x * 128 + wave * 16;
  const int col  = lane & 15;
  const bool hi  = lane >= 16;

  v8f acc[16] = {};
  for (int ks = 0; ks < CDIM; ks += 32) {
    __syncthreads();
    {
      const float* src = Wo + (size_t)tid * CDIM + ks;
      __bf16 tmp[32];
#pragma unroll
      for (int i = 0; i < 32; ++i) tmp[i] = (__bf16)src[i];
      uint4* dst = reinterpret_cast<uint4*>(&Wsl[tid * 32]);
      const uint4* s4 = reinterpret_cast<const uint4*>(tmp);
#pragma unroll
      for (int i = 0; i < 4; ++i) dst[i] = s4[i];
    }
    __syncthreads();
    const int kofs = ks + (hi ? 8 : 0);
    const __bf16* ap = aobuf + (size_t)(r0 + col) * CDIM + kofs;
    const v16bf a = ld_frag2(ap, ap + 16);
#pragma unroll
    for (int t = 0; t < 16; ++t) {
      const __bf16* wp = &Wsl[(t * 16 + col) * 32 + (hi ? 8 : 0)];
      acc[t] = wmma_bf16(a, ld_frag2(wp, wp + 16), acc[t]);
    }
  }
  // bias + residual
#pragma unroll
  for (int t = 0; t < 16; ++t) {
    const int c = t * 16 + col;
    const float bias = bo[c];
#pragma unroll
    for (int j = 0; j < 8; ++j) {
      const size_t rg = (size_t)(r0 + j + (hi ? 8 : 0));
      acc[t][j] += bias + (float)vres[rg * CDIM + c];
    }
  }
  // LayerNorm: reduce each row over 16 tiles (in-lane) then 16 lanes (shfl)
#pragma unroll
  for (int j = 0; j < 8; ++j) {
    float s = 0.f, sq = 0.f;
#pragma unroll
    for (int t = 0; t < 16; ++t) { const float v = acc[t][j]; s += v; sq += v * v; }
#pragma unroll
    for (int o = 1; o < 16; o <<= 1) { s += __shfl_xor(s, o, 32); sq += __shfl_xor(sq, o, 32); }
    const float mu   = s * (1.0f / CDIM);
    const float var  = sq * (1.0f / CDIM) - mu * mu;
    const float rstd = rsqrtf(var + 1e-6f);
    const size_t rg = (size_t)(r0 + j + (hi ? 8 : 0));
#pragma unroll
    for (int t = 0; t < 16; ++t) {
      const int c = t * 16 + col;
      out[rg * CDIM + c] = (acc[t][j] - mu) * rstd * gamma[c] + beta[c];
    }
  }
}

extern "C" void kernel_launch(void* const* d_in, const int* in_sizes, int n_in,
                              void* d_out, int out_size, void* d_ws, size_t ws_size,
                              hipStream_t stream) {
  (void)in_sizes; (void)n_in; (void)out_size; (void)ws_size;
  const float* x     = (const float*)d_in[0];
  const float* Wq    = (const float*)d_in[1];
  const float* bq    = (const float*)d_in[2];
  const float* Wk    = (const float*)d_in[3];
  const float* bk    = (const float*)d_in[4];
  const float* Wv    = (const float*)d_in[5];
  const float* bv    = (const float*)d_in[6];
  const float* Wo    = (const float*)d_in[7];
  const float* bo    = (const float*)d_in[8];
  const float* gamma = (const float*)d_in[9];
  const float* beta  = (const float*)d_in[10];
  float* out = (float*)d_out;

  __bf16* qbuf  = (__bf16*)d_ws;                                  // 8 MB
  __bf16* kbuf  = qbuf  + (size_t)NUM_B * NH * SEQ * DH;          // 8 MB
  __bf16* vbuf  = kbuf  + (size_t)NUM_B * NH * SEQ * DH;          // 8 MB (residual)
  __bf16* vtbuf = vbuf  + (size_t)MTOT * CDIM;                    // 8 MB (V^T per head)
  __bf16* aobuf = vtbuf + (size_t)NUM_B * NH * DH * SEQ;          // 8 MB

  qkv_proj_kernel<<<dim3(MTOT / 128, 3), 256, 0, stream>>>(
      x, Wq, bq, Wk, bk, Wv, bv, qbuf, kbuf, vbuf, vtbuf);
  flash_attn_kernel<<<dim3(SEQ / 128, NH, NUM_B), 256, 0, stream>>>(
      qbuf, kbuf, vtbuf, aobuf);
  oproj_ln_kernel<<<dim3(MTOT / 128), 256, 0, stream>>>(
      aobuf, Wo, bo, vbuf, gamma, beta, out);
}
